// ScannedRNN_12893491823157
// MI455X (gfx1250) — compile-verified, matrix operand order checked by reading, not run on hardware
//
#include <hip/hip_runtime.h>
#include <math.h>

// ---------------------------------------------------------------------------
// GRU scan for MI455X (gfx1250, wave32, WMMA).
// Persistent kernel: 32 WGs x 256 threads, device-wide split barrier per step.
// - 6 weight tiles (Wi/Wh x r/z/n, 512x16) live in LDS as bf16 hi/lo WMMA-B
//   fragments for the whole scan (ds_load_b128 per use).
// - Hidden state carried as (a) exact fp32 in d_out (= ys, carry term) and
//   (b) pre-masked bf16 hi/lo split scratch so h-side A fragments are pure
//   global_load_b128 -> v_wmma_f32_16x16x32_bf16 (no conversion VALU on the
//   recurrent critical path).
// - Input projection (ins@Wi) for step t+1 is computed in the BARRIER SHADOW:
//   arrive (atomicAdd) first, then do the independent ins-side GEMM, then
//   spin-wait on the generation counter. Its latency overlaps barrier
//   propagation + other WGs' work instead of serializing.
// - 3-term bf16 split (hi*hi + hi*lo + lo*hi, fp32 accum) ~ fp32 accuracy.
// ---------------------------------------------------------------------------

typedef __attribute__((ext_vector_type(16))) __bf16 v16bf;
typedef __attribute__((ext_vector_type(8)))  float  v8f;

#define T_STEPS  512
#define B_DIM    128
#define H_DIM    512
#define G3       1536          // 3*H
#define NWG      32            // workgroups (hidden-column groups of 16)
#define NTHREADS 256           // 8 waves -> 8 batch tiles of 16
#define NFRAG    (6*16*32)     // tiles * k-chunks * lanes
#define NFRAG_EL (NFRAG*16)    // bf16 elements per LDS buffer (96KB)

// d_ws layout
#define WS_BAR_CNT   0                       // unsigned
#define WS_BAR_GEN   32                      // unsigned (separate cacheline)
#define WS_HHI_OFF   1024                    // ushort[B*H] 128KB
#define WS_HLO_OFF   (1024 + B_DIM*H_DIM*2)  // ushort[B*H] 128KB

// K offset inside a 32-wide K chunk for fragment element e of a lane.
// 16-bit A/B fragment layout (ISA 7.12.2): lanes 0-15 hold K = e (e<8),
// 16+(e-8) (e>=8); lanes 16-31 add +8.
__device__ __forceinline__ int kmap(int e, int laneHi) {
  return ((e & 8) << 1) + (laneHi << 3) + (e & 7);
}

__device__ __forceinline__ v16bf ld_frag_lds(const unsigned short* s, int fidx) {
  union { uint4 q[2]; v16bf v; } u;
  const uint4* p = (const uint4*)(s + (size_t)fidx * 16);  // 32B aligned
  u.q[0] = p[0];
  u.q[1] = p[1];
  return u.v;
}

// A fragment straight from a pre-split bf16 row (two 16B loads, no VALU).
__device__ __forceinline__ v16bf ld_frag_bf16row(const unsigned short* rowPtr, int k0) {
  union { uint4 q[2]; v16bf v; } u;
  u.q[0] = *(const uint4*)(rowPtr + k0);        // e0..7  -> K = k0..k0+7
  u.q[1] = *(const uint4*)(rowPtr + k0 + 16);   // e8..15 -> K = k0+16..k0+23
  return u.v;
}

// Load 16 fp32 A-fragment elements for one 32-wide K chunk and split each
// into bf16 hi + bf16 lo (a ~= hi + lo).
__device__ __forceinline__ void load_a_frag_f32(const float* __restrict__ rowPtr,
                                                int kChunkBase, int laneHi,
                                                v16bf* hi, v16bf* lo) {
  const float* p0 = rowPtr + kChunkBase + (laneHi << 3);
  float4 a0 = *(const float4*)(p0 + 0);
  float4 a1 = *(const float4*)(p0 + 4);
  float4 a2 = *(const float4*)(p0 + 16);
  float4 a3 = *(const float4*)(p0 + 20);
  float v[16] = { a0.x, a0.y, a0.z, a0.w, a1.x, a1.y, a1.z, a1.w,
                  a2.x, a2.y, a2.z, a2.w, a3.x, a3.y, a3.z, a3.w };
#pragma unroll
  for (int e = 0; e < 16; ++e) {
    float f = v[e];
    __bf16 h = (__bf16)f;
    (*hi)[e] = h;
    (*lo)[e] = (__bf16)(f - (float)h);
  }
}

// acc += A*B with 3-term bf16 split: hi*hi + hi*lo + lo*hi (fp32 accumulate)
__device__ __forceinline__ v8f wmma3(v16bf aHi, v16bf aLo,
                                     const unsigned short* sHi,
                                     const unsigned short* sLo,
                                     int fidx, v8f acc) {
  v16bf bHi = ld_frag_lds(sHi, fidx);
  v16bf bLo = ld_frag_lds(sLo, fidx);
  acc = __builtin_amdgcn_wmma_f32_16x16x32_bf16(false, aHi, false, bHi, (short)0, acc, false, false);
  acc = __builtin_amdgcn_wmma_f32_16x16x32_bf16(false, aHi, false, bLo, (short)0, acc, false, false);
  acc = __builtin_amdgcn_wmma_f32_16x16x32_bf16(false, aLo, false, bHi, (short)0, acc, false, false);
  return acc;
}

// Input-side projection for one timestep: accumulates ins@Wi into r/z/xn.
__device__ __forceinline__ void ins_side(const float* __restrict__ insRow,
                                         int lane, int laneHi,
                                         const unsigned short* sBhi,
                                         const unsigned short* sBlo,
                                         v8f* aR, v8f* aZ, v8f* aXN) {
#pragma unroll 4
  for (int c = 0; c < 16; ++c) {
    v16bf aHi, aLo;
    load_a_frag_f32(insRow, 32 * c, laneHi, &aHi, &aLo);
    *aR  = wmma3(aHi, aLo, sBhi, sBlo, (0 * 16 + c) * 32 + lane, *aR);
    *aZ  = wmma3(aHi, aLo, sBhi, sBlo, (1 * 16 + c) * 32 + lane, *aZ);
    *aXN = wmma3(aHi, aLo, sBhi, sBlo, (2 * 16 + c) * 32 + lane, *aXN);
  }
}

__global__ __launch_bounds__(NTHREADS, 1)
void gru_scan_kernel(const float* __restrict__ ins,           // [T,B,H]
                     const unsigned char* __restrict__ resets,// [T,B] (bool)
                     const float* __restrict__ Wi,            // [H,3H]
                     const float* __restrict__ bi,            // [3H]
                     const float* __restrict__ Wh,            // [H,3H]
                     const float* __restrict__ b_hn,          // [H]
                     float* __restrict__ out,                 // [T,B,H] = ys
                     unsigned short* __restrict__ hHi,        // [B,H] bf16 hi (masked)
                     unsigned short* __restrict__ hLo,        // [B,H] bf16 lo (masked)
                     unsigned* barCnt, unsigned* barGen) {
  __shared__ __align__(16) unsigned short sBhi[NFRAG_EL];  // 96KB
  __shared__ __align__(16) unsigned short sBlo[NFRAG_EL];  // 96KB

  const int tid    = threadIdx.x;
  const int lane   = tid & 31;
  const int wave   = tid >> 5;          // batch tile 0..7
  const int laneHi = lane >> 4;
  const int laneN  = lane & 15;
  const int jBase  = blockIdx.x * 16;   // hidden-column base for this WG
  const int mBase  = wave * 16;         // batch-row base for this wave

  // ---- one-time: swizzle the 6 weight tiles (Wi/Wh x r/z/n) into LDS as
  // bf16 hi/lo B-fragments. Tile order: 0..2 = Wi(r,z,n), 3..5 = Wh(r,z,n).
  for (int f = tid; f < NFRAG; f += NTHREADS) {
    const int fl = f & 31;              // lane slot of fragment
    const int c  = (f >> 5) & 15;       // k chunk
    const int tt = f >> 9;              // tile 0..5
    const float* W = (tt < 3) ? Wi : Wh;
    const int col  = (tt % 3) * H_DIM + jBase + (fl & 15);
    const int flHi = fl >> 4;
#pragma unroll
    for (int e = 0; e < 16; ++e) {
      const int k = 32 * c + kmap(e, flHi);
      const float v = W[(size_t)k * G3 + col];
      union { __bf16 b; unsigned short u; } cvt;
      __bf16 h = (__bf16)v;
      cvt.b = h;                         sBhi[(size_t)f * 16 + e] = cvt.u;
      cvt.b = (__bf16)(v - (float)h);    sBlo[(size_t)f * 16 + e] = cvt.u;
    }
  }
  __syncthreads();

  const int rowA = mBase + laneN;       // batch row feeding this lane's A frag
  const int col  = jBase + laneN;       // this lane's output column
  const float biR = bi[col], biZ = bi[H_DIM + col], biN = bi[2 * H_DIM + col];
  const float bhn = b_hn[col];
  const unsigned short* hHiRow = hHi + (size_t)rowA * H_DIM;
  const unsigned short* hLoRow = hLo + (size_t)rowA * H_DIM;
  const v8f vzero = {0.f, 0.f, 0.f, 0.f, 0.f, 0.f, 0.f, 0.f};

  // Prologue: ins-side accumulators for t = 0
  v8f nR = vzero, nZ = vzero, nXN = vzero;
  ins_side(ins + (size_t)rowA * H_DIM, lane, laneHi, sBhi, sBlo, &nR, &nZ, &nXN);

#pragma unroll 1
  for (int t = 0; t < T_STEPS; ++t) {
    v8f accR = nR, accZ = nZ, accXN = nXN;   // ins@Wi contribution (pipelined)
    v8f accHN = vzero;

    const float* hPrev = out + (size_t)(t > 0 ? t - 1 : 0) * (B_DIM * H_DIM);
    const unsigned char* rst = resets + (size_t)t * B_DIM;

    if (t > 0) {
      // ---- recurrent GEMM: pre-split, pre-masked bf16 h -> pure loads + WMMA
#pragma unroll 4
      for (int c = 0; c < 16; ++c) {
        const int k0 = 32 * c + (laneHi << 3);
        v16bf hH = ld_frag_bf16row(hHiRow, k0);
        v16bf hL = ld_frag_bf16row(hLoRow, k0);
        accR  = wmma3(hH, hL, sBhi, sBlo, (3 * 16 + c) * 32 + lane, accR);
        accZ  = wmma3(hH, hL, sBhi, sBlo, (4 * 16 + c) * 32 + lane, accZ);
        accHN = wmma3(hH, hL, sBhi, sBlo, (5 * 16 + c) * 32 + lane, accHN);
      }
    }

    // ---- gates + state update (C/D layout: elem g -> row g+8*laneHi, col laneN)
    float* outT = out + (size_t)t * (B_DIM * H_DIM);
    const unsigned char* rstNext = resets + (size_t)(t + 1 < T_STEPS ? t + 1 : t) * B_DIM;

#pragma unroll
    for (int g = 0; g < 8; ++g) {
      const int row = mBase + g + (laneHi << 3);
      float hv = 0.f;
      if (t > 0) {
        const float mk = rst[row] ? 0.f : 1.f;      // mask h[t-1] with resets[t]
        hv = mk * hPrev[(size_t)row * H_DIM + col];
      }
      const float r  = 1.f / (1.f + __expf(-(accR[g] + biR)));
      const float z  = 1.f / (1.f + __expf(-(accZ[g] + biZ)));
      const float n  = tanhf(accXN[g] + biN + r * (accHN[g] + bhn));
      const float nh = (1.f - z) * n + z * hv;
      outT[(size_t)row * H_DIM + col] = nh;          // exact ys[t]
      if (t + 1 < T_STEPS) {
        // next step's A operand: masked by resets[t+1], bf16 hi/lo split
        const float nm = rstNext[row] ? 0.f : nh;
        union { __bf16 b; unsigned short u; } cvt;
        __bf16 hh = (__bf16)nm;
        cvt.b = hh;                          hHi[(size_t)row * H_DIM + col] = cvt.u;
        cvt.b = (__bf16)(nm - (float)hh);    hLo[(size_t)row * H_DIM + col] = cvt.u;
      }
    }

    // ---- split barrier: ARRIVE, then hide next step's ins GEMM in the
    //      barrier shadow, then WAIT.
    if (t + 1 < T_STEPS) {
      __threadfence();         // make h[t] stores device-visible
      __syncthreads();         // all waves of this WG fenced before arrive
      unsigned g0 = 0;
      if (tid == 0) {
        g0 = __hip_atomic_load(barGen, __ATOMIC_RELAXED, __HIP_MEMORY_SCOPE_AGENT);
        const unsigned a = __hip_atomic_fetch_add(barCnt, 1u, __ATOMIC_ACQ_REL, __HIP_MEMORY_SCOPE_AGENT);
        if (a == NWG - 1) {
          __hip_atomic_store(barCnt, 0u, __ATOMIC_RELAXED, __HIP_MEMORY_SCOPE_AGENT);
          __hip_atomic_fetch_add(barGen, 1u, __ATOMIC_RELEASE, __HIP_MEMORY_SCOPE_AGENT);
        }
      }

      // Barrier shadow: ins@Wi for step t+1 (independent of h[t]).
      nR = vzero; nZ = vzero; nXN = vzero;
      ins_side(ins + (size_t)(t + 1) * (B_DIM * H_DIM) + (size_t)rowA * H_DIM,
               lane, laneHi, sBhi, sBlo, &nR, &nZ, &nXN);
      if (t + 2 < T_STEPS)
        __builtin_prefetch(ins + (size_t)(t + 2) * (B_DIM * H_DIM) + (size_t)rowA * H_DIM, 0, 0);

      if (tid == 0) {
        while (__hip_atomic_load(barGen, __ATOMIC_ACQUIRE, __HIP_MEMORY_SCOPE_AGENT) == g0) {
          __builtin_amdgcn_s_sleep(1);
        }
      }
      __syncthreads();
      __threadfence();         // acquire: invalidate stale h lines
    }
  }
}

extern "C" void kernel_launch(void* const* d_in, const int* in_sizes, int n_in,
                              void* d_out, int out_size, void* d_ws, size_t ws_size,
                              hipStream_t stream) {
  (void)in_sizes; (void)n_in; (void)out_size; (void)ws_size;
  const float*         ins    = (const float*)d_in[0];
  const unsigned char* resets = (const unsigned char*)d_in[1];  // jnp.bool_ -> 1 byte
  const float*         Wi     = (const float*)d_in[2];
  const float*         bi     = (const float*)d_in[3];
  const float*         Wh     = (const float*)d_in[4];
  const float*         b_hn   = (const float*)d_in[5];
  float*               out    = (float*)d_out;

  unsigned char* ws = (unsigned char*)d_ws;
  unsigned*       barCnt = (unsigned*)(ws + WS_BAR_CNT);
  unsigned*       barGen = (unsigned*)(ws + WS_BAR_GEN);
  unsigned short* hHi    = (unsigned short*)(ws + WS_HHI_OFF);
  unsigned short* hLo    = (unsigned short*)(ws + WS_HLO_OFF);

  hipMemsetAsync(d_ws, 0, 1024, stream);  // zero barrier state (capture-legal)

  hipLaunchKernelGGL(gru_scan_kernel, dim3(NWG), dim3(NTHREADS), 0, stream,
                     ins, resets, Wi, bi, Wh, b_hn, out, hHi, hLo, barCnt, barGen);
}
